// CRAMModel_50268297232821
// MI455X (gfx1250) — matrix operand order repaired and checked by reading, not compiled, not run on hardware
//
#include <hip/hip_runtime.h>
#include <hip/hip_bf16.h>
#include <math.h>

typedef _Float16 v16h __attribute__((ext_vector_type(16)));
typedef _Float16 v8h  __attribute__((ext_vector_type(8)));
typedef _Float16 v4h  __attribute__((ext_vector_type(4)));
typedef float    v8f  __attribute__((ext_vector_type(8)));
typedef float    v4f  __attribute__((ext_vector_type(4)));

#define TBM 64
#define TBN 128
#define TBK 32
#define LDA_P 40   // halves per A row in LDS (80B pitch: 16B-aligned, conflict-free reads)
#define LDB_P 40   // halves per B(n-major) row in LDS

// ---------------- WMMA GEMM: C[M,N] = A[M,K] @ B, fp32 in/out, f16 matrix core ----
// transB==0: B is row-major [K,N].  transB==1: B is [N,K] (i.e. compute A @ B^T).
// Requires: M%64==0, N%128==0, K%32==0 (all shapes in this model comply).
__global__ __launch_bounds__(256)
void k_gemm(const float* __restrict__ A, const float* __restrict__ Bm,
            float* __restrict__ C, int M, int N, int K, int transB)
{
    __shared__ __align__(16) _Float16 As[TBM * LDA_P];
    __shared__ __align__(16) _Float16 Bs[TBN * LDB_P];

    const int tid  = threadIdx.x;
    const int lane = tid & 31;
    const int wave = tid >> 5;
    const int m0 = blockIdx.y * TBM;
    const int n0 = blockIdx.x * TBN;

    const int wm = (wave & 3) * 16;     // wave M sub-tile
    const int wn = (wave >> 2) * 64;    // wave N range (4 x 16)

    v8f acc[4];
#pragma unroll
    for (int j = 0; j < 4; ++j) acc[j] = (v8f)(0.0f);

    // A-fragment lane mapping (16x32 f16): lanes 0-15 -> K 0..7 / 16..23,
    // lanes 16-31 -> K 8..15 / 24..31, M = lane&15
    const int am    = lane & 15;
    const int abase = (lane >> 4) * 8;        // k offset of low 8 halves
    // B-fragment lane mapping (32x16 f16): lanes 0-15 hold K 0..7 / 16..23 (N=lane),
    // lanes 16-31 hold K 8..15 / 24..31 (N=lane-16)
    const int bn    = lane & 15;
    const int bkoff = (lane >> 4) * 8;

    // straight-line staging coordinates (blockDim.x == 256 exactly)
    const int s_ar = tid >> 2;           // A row 0..63
    const int s_ac = (tid & 3) * 8;      // A col 0,8,16,24
    const int s_kb = (tid >> 5) * 4;     // B k-block 0..28 (row-major B)
    const int s_nb = (tid & 31) * 4;     // B n-block 0..124 (row-major B)
    const int s_tk = (tid & 7) * 4;      // B k-block (transB)
    const int s_tn = tid >> 3;           // B n-row 0..31 (transB, +j*32)

    for (int kk = 0; kk < K; kk += TBK) {
        __syncthreads();

        // ---- stage A tile (64x32 fp32 -> f16), 2x b128 load + 1x b128 LDS store ----
        {
            const float* src = &A[(size_t)(m0 + s_ar) * K + kk + s_ac];
            v4f f0 = *(const v4f*)src;
            v4f f1 = *(const v4f*)(src + 4);
            v8h h;
#pragma unroll
            for (int i = 0; i < 4; ++i) { h[i] = (_Float16)f0[i]; h[4 + i] = (_Float16)f1[i]; }
            *(v8h*)&As[s_ar * LDA_P + s_ac] = h;
        }

        // ---- stage B tile as Bs[n][k] (n-major so fragments are contiguous) --------
        if (transB) {
            // B is [N,K]: read 16B along K per n-row; contiguous global + LDS
#pragma unroll
            for (int j = 0; j < 4; ++j) {
                const int n = s_tn + j * 32;
                const float* src = &Bm[(size_t)(n0 + n) * K + kk + s_tk];
                v4f f = *(const v4f*)src;
                v4h h;
#pragma unroll
                for (int i = 0; i < 4; ++i) h[i] = (_Float16)f[i];
                *(v4h*)&Bs[n * LDB_P + s_tk] = h;
            }
        } else {
            // B is [K,N]: 4x4 sub-block; coalesced row reads, register transpose
            const float* src = &Bm[(size_t)(kk + s_kb) * N + n0 + s_nb];
            v4f f0 = *(const v4f*)src;
            v4f f1 = *(const v4f*)(src + N);
            v4f f2 = *(const v4f*)(src + 2 * (size_t)N);
            v4f f3 = *(const v4f*)(src + 3 * (size_t)N);
#pragma unroll
            for (int n = 0; n < 4; ++n) {
                v4h h;
                h[0] = (_Float16)f0[n]; h[1] = (_Float16)f1[n];
                h[2] = (_Float16)f2[n]; h[3] = (_Float16)f3[n];
                *(v4h*)&Bs[(s_nb + n) * LDB_P + s_kb] = h;
            }
        }

        if (kk + TBK < K && tid < TBM)
            __builtin_prefetch(&A[(size_t)(m0 + tid) * K + kk + TBK], 0, 1);
        __syncthreads();

        const _Float16* ap = &As[(wm + am) * LDA_P + abase];
        v8h alo = *(const v8h*)ap;
        v8h ahi = *(const v8h*)(ap + 16);
        v16h afrag;
#pragma unroll
        for (int i = 0; i < 8; ++i) { afrag[i] = alo[i]; afrag[8 + i] = ahi[i]; }

#pragma unroll
        for (int j = 0; j < 4; ++j) {
            const _Float16* bp = &Bs[(wn + j * 16 + bn) * LDB_P + bkoff];
            v8h blo = *(const v8h*)bp;
            v8h bhi = *(const v8h*)(bp + 16);
            v16h bfrag;
#pragma unroll
            for (int i = 0; i < 8; ++i) { bfrag[i] = blo[i]; bfrag[8 + i] = bhi[i]; }
            acc[j] = __builtin_amdgcn_wmma_f32_16x16x32_f16(
                false, afrag, false, bfrag, (short)0, acc[j], false, false);
        }
    }

    // C layout: acc element r -> M = wm + r + (lane>=16 ? 8 : 0), N = base + (lane&15)
    const int cm = m0 + wm + (lane >> 4) * 8;
#pragma unroll
    for (int j = 0; j < 4; ++j) {
        const int cn = n0 + wn + j * 16 + (lane & 15);
#pragma unroll
        for (int r = 0; r < 8; ++r)
            C[(size_t)(cm + r) * N + cn] = acc[j][r];
    }
}

// ---------------- small-N GEMV: C[M,N] = A[M,K] @ B[K,N], N <= 8 -------------------
__global__ __launch_bounds__(256)
void k_gemv_small(const float* __restrict__ A, const float* __restrict__ Bm,
                  float* __restrict__ C, int N, int K)
{
    __shared__ float red[256];
    const int m = blockIdx.x, t = threadIdx.x;
    for (int n = 0; n < N; ++n) {
        float s = 0.f;
        for (int k = t; k < K; k += 256)
            s += A[(size_t)m * K + k] * Bm[(size_t)k * N + n];
        red[t] = s; __syncthreads();
        for (int off = 128; off > 0; off >>= 1) {
            if (t < off) red[t] += red[t + off];
            __syncthreads();
        }
        if (t == 0) C[(size_t)m * N + n] = red[0];
        __syncthreads();
    }
}

// ---------------- rmsnorm: out = x * w * rsqrt(mean(x^2)+eps) ----------------------
__global__ __launch_bounds__(256)
void k_rmsnorm(const float* __restrict__ x, const float* __restrict__ w,
               float* __restrict__ o, int D)
{
    __shared__ float red[256];
    const int r = blockIdx.x, t = threadIdx.x;
    float s = 0.f;
    for (int d = t; d < D; d += 256) { float v = x[(size_t)r * D + d]; s += v * v; }
    red[t] = s; __syncthreads();
    for (int off = 128; off > 0; off >>= 1) {
        if (t < off) red[t] += red[t + off];
        __syncthreads();
    }
    const float inv = rsqrtf(red[0] / D + 1e-6f);
    for (int d = t; d < D; d += 256)
        o[(size_t)r * D + d] = x[(size_t)r * D + d] * w[d] * inv;
}

// residual + gated rmsnorm epilogue:
// out = base + gate(row)*beta * (h1-h2)*w*rsqrt(mean((h1-h2)^2)+eps)
__global__ __launch_bounds__(256)
void k_rgn(float* __restrict__ out, const float* __restrict__ base,
           const float* __restrict__ h1, const float* __restrict__ h2,
           const float* __restrict__ w, const float* __restrict__ beta,
           const float* __restrict__ gate, int gstride, int goff, int D)
{
    __shared__ float red[256];
    const int r = blockIdx.x, t = threadIdx.x;
    float ss = 0.f;
    for (int d = t; d < D; d += 256) {
        size_t i = (size_t)r * D + d;
        float v = h1[i] - (h2 ? h2[i] : 0.f);
        ss += v * v;
    }
    red[t] = ss; __syncthreads();
    for (int off = 128; off > 0; off >>= 1) {
        if (t < off) red[t] += red[t + off];
        __syncthreads();
    }
    const float inv = rsqrtf(red[0] / D + 1e-6f);
    const float g = gate ? gate[(size_t)r * gstride + goff] : 1.f;
    const float bs = g * beta[0];
    for (int d = t; d < D; d += 256) {
        size_t i = (size_t)r * D + d;
        float v = h1[i] - (h2 ? h2[i] : 0.f);
        out[i] = base[i] + bs * v * w[d] * inv;
    }
}

// ---------------- softmax (tiny cols, thread-0 serial, in place) -------------------
__global__ void k_softmax_small(float* __restrict__ x, int cols)
{
    if (threadIdx.x != 0) return;
    const size_t b = (size_t)blockIdx.x * cols;
    float mx = -1e30f;
    for (int c = 0; c < cols; ++c) mx = fmaxf(mx, x[b + c]);
    float s = 0.f;
    for (int c = 0; c < cols; ++c) { float e = __expf(x[b + c] - mx); x[b + c] = e; s += e; }
    for (int c = 0; c < cols; ++c) x[b + c] /= s;
}

// softmax over cols==256 rows (scale applied pre-softmax), in place
__global__ __launch_bounds__(256)
void k_softmax_block(float* __restrict__ x, int cols, float scale)
{
    __shared__ float red[256];
    const int r = blockIdx.x, t = threadIdx.x;
    const size_t b = (size_t)r * cols;
    float v = x[b + t] * scale;
    red[t] = v; __syncthreads();
    for (int off = 128; off > 0; off >>= 1) {
        if (t < off) red[t] = fmaxf(red[t], red[t + off]);
        __syncthreads();
    }
    const float mx = red[0]; __syncthreads();
    const float e = __expf(v - mx);
    red[t] = e; __syncthreads();
    for (int off = 128; off > 0; off >>= 1) {
        if (t < off) red[t] += red[t + off];
        __syncthreads();
    }
    x[b + t] = e / red[0];
}

// ---------------- RSP recurrent scan (serial over L) -------------------------------
__global__ __launch_bounds__(256)
void k_scan(const float* __restrict__ u, const float* __restrict__ gf,
            const float* __restrict__ gs, float* __restrict__ y, int L, int Cd)
{
    const int b = blockIdx.x;
    const int c = blockIdx.y * blockDim.x + threadIdx.x;
    size_t base = ((size_t)b * L) * Cd + c;
    float f = 0.f, s = 0.f;
    for (int l = 0; l < L; ++l) {
        size_t i = base + (size_t)l * Cd;
        float uv = u[i];
        float a = 1.f / (1.f + __expf(-gf[i]));
        float g = 1.f / (1.f + __expf(-gs[i]));
        f = a * f + (1.f - a) * uv;
        s = g * s + (1.f - g) * f;
        y[i] = f * (s / (1.f + __expf(-s)));   // f * silu(s)
    }
}

// ---------------- elementwise helpers ----------------------------------------------
__global__ void k_silu_mul(float* a, const float* b, size_t n) {
    size_t i = (size_t)blockIdx.x * blockDim.x + threadIdx.x;
    if (i < n) { float v = a[i]; a[i] = v / (1.f + __expf(-v)) * b[i]; }
}
__global__ void k_tanh_k(float* a, size_t n) {
    size_t i = (size_t)blockIdx.x * blockDim.x + threadIdx.x;
    if (i < n) a[i] = tanhf(a[i]);
}
__global__ void k_add(float* a, const float* b, size_t n) {
    size_t i = (size_t)blockIdx.x * blockDim.x + threadIdx.x;
    if (i < n) a[i] += b[i];
}
__global__ void k_copy(float* o, const float* x, size_t n) {
    size_t i = (size_t)blockIdx.x * blockDim.x + threadIdx.x;
    if (i < n) o[i] = x[i];
}
__global__ void k_zero_f(float* p, size_t n) {
    size_t i = (size_t)blockIdx.x * blockDim.x + threadIdx.x;
    if (i < n) p[i] = 0.f;
}
__global__ void k_zero_i(int* p, int n) {
    int i = blockIdx.x * blockDim.x + threadIdx.x;
    if (i < n) p[i] = 0;
}

__global__ void k_embed(const int* __restrict__ ids, const float* __restrict__ emb,
                        float* __restrict__ x, int D) {
    const int t0 = blockIdx.x;
    const int id = ids[t0];
    for (int d = threadIdx.x; d < D; d += 256)
        x[(size_t)t0 * D + d] = emb[(size_t)id * D + d] * 32.0f;  // sqrt(1024)
}

// h_rsp mixture: (p_fast+p_mem+p_rsn)*h_fast + p_deep*h_deep + p_wide*h_wide
__global__ __launch_bounds__(256)
void k_combine_rsp(const float* hf, const float* hd, const float* hw,
                   const float* route, float* out, int D) {
    const int r = blockIdx.x;
    const float cf = route[r * 5 + 0] + route[r * 5 + 3] + route[r * 5 + 4];
    const float pd = route[r * 5 + 1], pw = route[r * 5 + 2];
    for (int d = threadIdx.x; d < D; d += 256) {
        size_t i = (size_t)r * D + d;
        out[i] = cf * hf[i] + pd * hd[i] + pw * hw[i];
    }
}

// h_am = h + p_mem * sigmoid(gvec[row]) * read   (h_mem - h == gate*read)
__global__ __launch_bounds__(256)
void k_samg(float* out, const float* h, const float* rd, const float* gvec,
            const float* route, int D) {
    const int r = blockIdx.x;
    const float c = route[r * 5 + 3] / (1.f + __expf(-gvec[r]));
    for (int d = threadIdx.x; d < D; d += 256) {
        size_t i = (size_t)r * D + d;
        out[i] = h[i] + c * rd[i];
    }
}

// MoE: top-2 flags (any token) + weighted accumulation
__global__ void k_moe_flags(const float* probs, int rows, int* flags) {
    int r = blockIdx.x * blockDim.x + threadIdx.x;
    if (r >= rows) return;
    const float* p = probs + (size_t)r * 4;
    int b0 = 0; float v0 = p[0];
    for (int e = 1; e < 4; ++e) if (p[e] > v0) { v0 = p[e]; b0 = e; }
    int b1 = -1; float v1 = -1e30f;
    for (int e = 0; e < 4; ++e) if (e != b0 && p[e] > v1) { v1 = p[e]; b1 = e; }
    atomicOr(&flags[b0], 1); atomicOr(&flags[b1], 1);
}
__global__ __launch_bounds__(256)
void k_moe_acc(float* out, const float* eo, const float* probs,
               const int* flags, int e, int D) {
    const int r = blockIdx.x;
    const float w = flags[e] ? probs[(size_t)r * 4 + e] : 0.f;
    for (int d = threadIdx.x; d < D; d += 256) {
        size_t i = (size_t)r * D + d;
        out[i] += w * eo[i];
    }
}

// =====================================================================================
extern "C" void kernel_launch(void* const* d_in, const int* in_sizes, int n_in,
                              void* d_out, int out_size, void* d_ws, size_t ws_size,
                              hipStream_t stream)
{
    (void)in_sizes; (void)n_in; (void)out_size; (void)ws_size;
    const int M = 4096, D = 1024, V = 32000, NL = 4;
    const size_t SZ = (size_t)M * D;

    // JAX sorted-key pytree flatten order:
    // 0: input_ids, 1: emb, then per layer (38 leaves):
    //  +0 adr.wd +1 adr.wr | +2 ffn.w1 +3 ffn.w2 +4 ffn.w3
    //  +5+3e moe.e[e].w1 +6+3e .w2 +7+3e .w3 (e=0..3) | +17 moe.router
    //  +18..20 norm_ffn{beta,w_out,w_pre} | +21..23 norm_rsp | +24..26 norm_sle
    //  +27..30 rsp_deep{wgf,wgs,wo,wu} | +31..34 rsp_fast{wgf,wgs,wo,wu}
    //  +35 sle.w1 +36 sle.w2 +37 sle.wc
    // 154 samg.mem, 155 samg.wg, 156 w_final
    const int*   ids = (const int*)d_in[0];
    const float* emb = (const float*)d_in[1];
    const float* mem = (const float*)d_in[2 + NL * 38 + 0];
    const float* wg  = (const float*)d_in[2 + NL * 38 + 1];
    const float* wfin= (const float*)d_in[2 + NL * 38 + 2];

    float* ws = (float*)d_ws;
    float* A_  = ws + 0 * SZ;    // x (persistent)
    float* B_  = ws + 1 * SZ;    // xn / ffn-norm
    float* C_  = ws + 2 * SZ;    // u / sle t / ffn f2 scratch
    float* Dd_ = ws + 3 * SZ;    // gf / sle t2
    float* E_  = ws + 4 * SZ;    // gs
    float* F_  = ws + 5 * SZ;    // scan y
    float* G_  = ws + 6 * SZ;    // h_fast / ffn out
    float* H_  = ws + 7 * SZ;    // h_deep
    float* I_  = ws + 8 * SZ;    // expert out
    float* J_  = ws + 9 * SZ;    // h_wide
    float* K_  = ws + 10 * SZ;   // h_rsp
    float* L_  = ws + 11 * SZ;   // samg read
    float* Mb_ = ws + 12 * SZ;   // h_am
    float* Nb_ = ws + 13 * SZ;   // sle hp
    float* big1 = C_;            // overlays C..F (dead during MoE/FFN)
    float* big2 = K_;            // overlays K..Nb (dead during MoE/FFN)
    float* small = ws + 14 * SZ;
    float* route  = small;                    // 4096*5
    float* probs  = small + 32768;            // 4096*4
    float* scores = small + 65536;            // 4096*256
    float* gvec   = small + 65536 + 1048576;  // 4096
    int*   flags  = (int*)(small + 65536 + 1048576 + 8192);

    auto gemm = [&](const float* Ax, const float* Bx, float* Cx,
                    int Mx, int Nx, int Kx, int tb) {
        dim3 g(Nx / TBN, Mx / TBM);
        k_gemm<<<g, dim3(256), 0, stream>>>(Ax, Bx, Cx, Mx, Nx, Kx, tb);
    };
    auto eb = [](size_t n) { return dim3((unsigned)((n + 255) / 256)); };

    // x = emb[ids] * sqrt(D)
    k_embed<<<M, 256, 0, stream>>>(ids, emb, A_, D);

    for (int l = 0; l < NL; ++l) {
        const int lb = 2 + l * 38;
        auto LP = [&](int i) { return (const float*)d_in[lb + i]; };

        // xn = rmsnorm(x, w_pre_rsp)
        k_rmsnorm<<<M, 256, 0, stream>>>(A_, LP(23), B_, D);
        // route = softmax(xn @ wr)
        k_gemv_small<<<M, 256, 0, stream>>>(B_, LP(1), route, 5, D);
        k_softmax_small<<<M, 32, 0, stream>>>(route, 5);
        // MoE router probs + top2 selection flags
        k_gemv_small<<<M, 256, 0, stream>>>(B_, LP(17), probs, 4, D);
        k_softmax_small<<<M, 32, 0, stream>>>(probs, 4);
        k_zero_i<<<1, 32, 0, stream>>>(flags, 4);
        k_moe_flags<<<(M + 255) / 256, 256, 0, stream>>>(probs, M, flags);

        // rsp_fast
        gemm(B_, LP(34), C_, M, 1024, D, 0);   // u   = xn @ wu
        gemm(B_, LP(31), Dd_, M, 1024, D, 0);  // gf  = xn @ wgf
        gemm(B_, LP(32), E_, M, 1024, D, 0);   // gs  = xn @ wgs
        k_scan<<<dim3(4, 4), 256, 0, stream>>>(C_, Dd_, E_, F_, 1024, 1024);
        gemm(F_, LP(33), G_, M, D, 1024, 0);   // h_fast = y @ wo
        // rsp_deep
        gemm(B_, LP(30), C_, M, 1024, D, 0);
        gemm(B_, LP(27), Dd_, M, 1024, D, 0);
        gemm(B_, LP(28), E_, M, 1024, D, 0);
        k_scan<<<dim3(4, 4), 256, 0, stream>>>(C_, Dd_, E_, F_, 1024, 1024);
        gemm(F_, LP(29), H_, M, D, 1024, 0);   // h_deep

        // MoE wide path
        k_zero_f<<<eb(SZ), 256, 0, stream>>>(J_, SZ);
        for (int e = 0; e < 4; ++e) {
            gemm(B_, LP(5 + 3 * e), big1, M, 2048, D, 0);
            gemm(B_, LP(7 + 3 * e), big2, M, 2048, D, 0);
            k_silu_mul<<<eb((size_t)M * 2048), 256, 0, stream>>>(big1, big2, (size_t)M * 2048);
            gemm(big1, LP(6 + 3 * e), I_, M, D, 2048, 0);
            k_moe_acc<<<M, 256, 0, stream>>>(J_, I_, probs, flags, e, D);
        }

        // mix paths + residual rmsnorm:  h_rsp = x + beta*rmsnorm(mix, w_out)
        k_combine_rsp<<<M, 256, 0, stream>>>(G_, H_, J_, route, K_, D);
        k_rgn<<<M, 256, 0, stream>>>(K_, A_, K_, nullptr, LP(22), LP(21),
                                     nullptr, 0, 0, D);

        // SAMG read: scores = softmax(h @ mem^T / 32); read = scores @ mem
        gemm(K_, mem, scores, M, 256, D, 1);
        k_softmax_block<<<M, 256, 0, stream>>>(scores, 256, 1.0f / 32.0f);
        gemm(scores, mem, L_, M, D, 256, 0);
        k_gemv_small<<<M, 256, 0, stream>>>(K_, wg, gvec, 1, D);
        k_samg<<<M, 256, 0, stream>>>(Mb_, K_, L_, gvec, route, D);  // h_am

        // SLE: hp = h_am; 2x { hp += tanh(hp@w1) @ w2 }
        k_copy<<<eb(SZ), 256, 0, stream>>>(Nb_, Mb_, SZ);
        for (int it = 0; it < 2; ++it) {
            gemm(Nb_, LP(35), C_, M, D, D, 0);
            k_tanh_k<<<eb(SZ), 256, 0, stream>>>(C_, SZ);
            gemm(C_, LP(36), Dd_, M, D, D, 0);
            k_add<<<eb(SZ), 256, 0, stream>>>(Nb_, Dd_, SZ);
        }
        // h_as = h_am + p_rsn*beta_sle*rmsnorm(hp - h_am, w_out_sle) -> into A_
        k_rgn<<<M, 256, 0, stream>>>(A_, Mb_, Nb_, Mb_, LP(25), LP(24),
                                     route, 5, 4, D);

        // FFN: h_out = h_as + beta*rmsnorm(swiglu(rmsnorm(h_as)), w_out)
        k_rmsnorm<<<M, 256, 0, stream>>>(A_, LP(20), B_, D);
        gemm(B_, LP(2), big1, M, 4096, D, 0);
        gemm(B_, LP(4), big2, M, 4096, D, 0);
        k_silu_mul<<<eb((size_t)M * 4096), 256, 0, stream>>>(big1, big2, (size_t)M * 4096);
        gemm(big1, LP(3), G_, M, D, 4096, 0);
        k_rgn<<<M, 256, 0, stream>>>(A_, A_, G_, nullptr, LP(19), LP(18),
                                     nullptr, 0, 0, D);
    }

    // final: logits = rmsnorm(x, w_final) @ emb^T
    k_rmsnorm<<<M, 256, 0, stream>>>(A_, wfin, B_, D);
    gemm(B_, emb, (float*)d_out, M, V, D, 1);
}
